// UnMaskedHead_27049704030328
// MI455X (gfx1250) — matrix operand order repaired and checked by reading, not compiled, not run on hardware
//
#include <hip/hip_runtime.h>
#include <hip/hip_bf16.h>

// ---------------------------------------------------------------------------
// UnMaskedHead on MI455X (gfx1250, wave32, WMMA bf16)
// B=8, T=2048, D=1024, HS=64
// Roofline: rel_pos stream = 1 GB fp32 @ 23.3 TB/s => ~43us floor; all matmuls
// (19 GFLOP) are far below that, so the design reads rel_pos exactly once and
// maximizes the number of independent HBM streams (SPLIT_V * T/16 = 512 WGs).
// ---------------------------------------------------------------------------

#define B_  8
#define T_  2048
#define D_  1024
#define HS_ 64
#define SPLIT_V 4                       // key-dim split for occupancy
#define VCHUNK  (T_ / SPLIT_V)          // 512 keys per workgroup

typedef __attribute__((ext_vector_type(16))) __bf16 v16bf;
typedef __attribute__((ext_vector_type(8)))  __bf16 v8bf;
typedef __attribute__((ext_vector_type(8)))  float  v8f;

static __device__ inline v8f wmma_bf16(v16bf a, v16bf b, v8f c) {
  // v_wmma_f32_16x16x32_bf16: (neg_a, A, neg_b, B, c_mod, C, reuse_a, reuse_b)
  return __builtin_amdgcn_wmma_f32_16x16x32_bf16(false, a, false, b, (short)0, c,
                                                 false, false);
}

static __device__ inline v8f zero_v8f() {
  v8f z;
#pragma unroll
  for (int i = 0; i < 8; ++i) z[i] = 0.0f;
  return z;
}

static __device__ inline v16bf zero_v16bf() {
  v16bf z;
#pragma unroll
  for (int i = 0; i < 16; ++i) z[i] = (__bf16)0.0f;
  return z;
}

// A-matrix (16x32, 16-bit) per-lane fragment: elements 0..7 <- p0[0..7],
// elements 8..15 <- p1[0..7].  (K runs {16h..16h+7} and {16h+8..16h+15}.)
static __device__ inline v16bf load_A_bf16(const __bf16* p0, const __bf16* p1) {
  v8bf a = *(const v8bf*)p0;
  v8bf b = *(const v8bf*)p1;
  v16bf r;
#pragma unroll
  for (int i = 0; i < 8; ++i) { r[i] = a[i]; r[8 + i] = b[i]; }
  return r;
}

// B-matrix (32x16, 16-bit) per-lane fragment from 16 consecutive fp32 values
// (element e <-> K = 16*half + e).
static __device__ inline v16bf load_B16_f32(const float* p) {
  v16bf r;
#pragma unroll
  for (int e = 0; e < 16; ++e) r[e] = (__bf16)p[e];
  return r;
}

// ---------------------------------------------------------------------------
// Kernel 1: projection  val[m,n] = (x[m,:] . W[n,:] + bias[n]) * scale  (bf16)
// grid.x = (B*T)/16 m-tiles, block = 128 (4 waves; wave w owns n-tile w).
// WRITE_T=false: writes out[m*HS+n].  WRITE_T=true: writes the transposed
// copy outT[(b*HS+n)*T + t] (used for V so PV B-fragments load contiguously).
// ---------------------------------------------------------------------------
template <bool WRITE_T>
__global__ __launch_bounds__(128) void proj_kernel(
    const float* __restrict__ x, const float* __restrict__ W,
    const float* __restrict__ bias, __bf16* __restrict__ dst, float scale) {
  const int wave = threadIdx.x >> 5;     // 0..3 -> n-tile
  const int lane = threadIdx.x & 31;
  const int ln   = lane & 15;            // A: row m / B: col n / C: col n
  const int h    = lane >> 4;            // lane half
  const int mbase = blockIdx.x * 16;
  const int nbase = wave * 16;

  v8f acc = zero_v8f();
  const float* xr = x + (size_t)(mbase + ln) * D_;
  const float* wr = W + (size_t)(nbase + ln) * D_;

  for (int k0 = 0; k0 < D_; k0 += 32) {
    v16bf a;
#pragma unroll
    for (int e = 0; e < 8; ++e) {
      a[e]     = (__bf16)xr[k0 + 16 * h + e];
      a[8 + e] = (__bf16)xr[k0 + 16 * h + 8 + e];
    }
    v16bf b = load_B16_f32(wr + k0 + 16 * h);   // B[c,n] = W[n,c]
    acc = wmma_bf16(a, b, acc);
  }

  const float bval = bias[nbase + ln];
#pragma unroll
  for (int r = 0; r < 8; ++r) {
    const int M = mbase + r + 8 * h;                 // flat row = b*T + t
    const float val = (acc[r] + bval) * scale;
    if constexpr (WRITE_T) {
      const int bb = M >> 11;                        // T = 2048
      const int tt = M & (T_ - 1);
      dst[((size_t)(bb * HS_ + nbase + ln)) * T_ + tt] = (__bf16)val;
    } else {
      dst[(size_t)M * HS_ + nbase + ln] = (__bf16)val;
    }
  }
}

// ---------------------------------------------------------------------------
// Kernel 2: flash attention with relative-position bias (partial over keys).
// grid = (T/16 t-tiles, SPLIT_V key-splits).  block = 256 = 8 waves;
// wave w = batch b=w, so each rel_pos chunk is loaded ONCE per WG and shared
// across all 8 batches through LDS.  Emits un-normalized (acc, m, l).
// ---------------------------------------------------------------------------
__global__ __launch_bounds__(256) void attn_kernel(
    const __bf16* __restrict__ qB,    // [B,T,HS] bf16 (raw q)
    const __bf16* __restrict__ kB,    // [B,T,HS] bf16 (8*k, sqrt(HS) folded)
    const __bf16* __restrict__ vBt,   // [B,HS,T] bf16 (v transposed)
    const float*  __restrict__ rel,   // [T,T,HS] fp32
    float* __restrict__ pacc,         // [SPLIT_V,B,T,HS]
    float* __restrict__ pm,           // [SPLIT_V,B,T]
    float* __restrict__ pl) {         // [SPLIT_V,B,T]
  __shared__ float  lds_rel[16][B_][32];        // [t_local][b][v_local] 16 KB
  __shared__ __bf16 lds_p[B_][16][32];          // per-wave P tile        8 KB

  const int wave = threadIdx.x >> 5;   // batch index
  const int lane = threadIdx.x & 31;
  const int ln   = lane & 15;
  const int h    = lane >> 4;
  const int tbase = blockIdx.x * 16;
  const int split = blockIdx.y;
  const int voff  = split * VCHUNK;

  // Q fragment for the score matmul: A[m=t_local, c], K split 0..31 / 32..63
  const __bf16* qrow = qB + ((size_t)wave * T_ + tbase + ln) * HS_;
  const v16bf aq0 = load_A_bf16(qrow + 16 * h,      qrow + 8 + 16 * h);
  const v16bf aq1 = load_A_bf16(qrow + 32 + 16 * h, qrow + 40 + 16 * h);

  // Q fragments for the rel-bias matmuls: per assigned t (2 per wave),
  // A rows = batches (rows 8..15 zero).
  v16bf abt[2][2];
#pragma unroll
  for (int i = 0; i < 2; ++i) {
    const int tg = tbase + 2 * wave + i;
    if (ln < B_) {
      const __bf16* r = qB + ((size_t)ln * T_ + tg) * HS_;
      abt[i][0] = load_A_bf16(r + 16 * h,      r + 8 + 16 * h);
      abt[i][1] = load_A_bf16(r + 32 + 16 * h, r + 40 + 16 * h);
    } else {
      abt[i][0] = zero_v16bf();
      abt[i][1] = zero_v16bf();
    }
  }

  v8f acc[4];
#pragma unroll
  for (int nt = 0; nt < 4; ++nt) acc[nt] = zero_v8f();
  float mrow[8], lrow[8];
#pragma unroll
  for (int r = 0; r < 8; ++r) { mrow[r] = -3.0e38f; lrow[r] = 0.0f; }

  for (int vb = voff; vb < voff + VCHUNK; vb += 32) {
    // ---- rel bias for this wave's two assigned t's (shared across batches)
#pragma unroll
    for (int i = 0; i < 2; ++i) {
      const int tloc = 2 * wave + i;
      const int tg   = tbase + tloc;
      const float* Rbase = rel + ((size_t)tg * T_ + vb) * HS_;
#pragma unroll
      for (int vt = 0; vt < 2; ++vt) {
        const float* p0 = Rbase + (size_t)(vt * 16 + ln) * HS_ + 16 * h;
        __builtin_prefetch((const void*)(p0 + 32 * (size_t)HS_), 0, 0);
        v16bf b0 = load_B16_f32(p0);        // c = 0..31
        v16bf b1 = load_B16_f32(p0 + 32);   // c = 32..63
        v8f d = wmma_bf16(abt[i][0], b0, wmma_bf16(abt[i][1], b1, zero_v8f()));
        if (h == 0) {                        // rows 0..7 = batches
#pragma unroll
          for (int r = 0; r < 8; ++r) lds_rel[tloc][r][vt * 16 + ln] = d[r];
        }
      }
    }
    __syncthreads();

    // ---- scores for own batch + rel bias
    v8f s[2];
#pragma unroll
    for (int vt = 0; vt < 2; ++vt) {
      const __bf16* kp =
          kB + ((size_t)wave * T_ + vb + vt * 16 + ln) * HS_ + 16 * h;
      v16bf bk0 = *(const v16bf*)kp;          // c = 0..31
      v16bf bk1 = *(const v16bf*)(kp + 32);   // c = 32..63
      s[vt] = wmma_bf16(aq0, bk0, wmma_bf16(aq1, bk1, zero_v8f()));
#pragma unroll
      for (int r = 0; r < 8; ++r)
        s[vt][r] += lds_rel[r + 8 * h][wave][vt * 16 + ln];
    }

    // ---- online softmax over the 32 new columns
#pragma unroll
    for (int r = 0; r < 8; ++r) {
      float mx = fmaxf(s[0][r], s[1][r]);
#pragma unroll
      for (int msk = 1; msk < 16; msk <<= 1)
        mx = fmaxf(mx, __shfl_xor(mx, msk, 32));
      const float mn = fmaxf(mrow[r], mx);
      const float alpha = __expf(mrow[r] - mn);
      mrow[r] = mn;
      const float p0 = __expf(s[0][r] - mn);
      const float p1 = __expf(s[1][r] - mn);
      float rs = p0 + p1;
#pragma unroll
      for (int msk = 1; msk < 16; msk <<= 1) rs += __shfl_xor(rs, msk, 32);
      lrow[r] = lrow[r] * alpha + rs;
#pragma unroll
      for (int nt = 0; nt < 4; ++nt) acc[nt][r] *= alpha;
      lds_p[wave][r + 8 * h][ln]      = (__bf16)p0;   // C-layout -> LDS
      lds_p[wave][r + 8 * h][16 + ln] = (__bf16)p1;
    }

    // ---- P back in A-layout (same-wave LDS round trip), then PV
    const __bf16* pr = &lds_p[wave][ln][0];
    v16bf ap = load_A_bf16(pr + 16 * h, pr + 8 + 16 * h);
#pragma unroll
    for (int nt = 0; nt < 4; ++nt) {
      const __bf16* vp =
          vBt + ((size_t)(wave * HS_ + nt * 16 + ln)) * T_ + vb + 16 * h;
      v16bf bv = *(const v16bf*)vp;           // K = v_local contiguous
      acc[nt] = wmma_bf16(ap, bv, acc[nt]);
    }
    __syncthreads();   // protect lds_rel for next iteration
  }

  // ---- epilogue: store un-normalized partials for this key split
#pragma unroll
  for (int r = 0; r < 8; ++r) {
    const size_t row = (size_t)split * B_ * T_ + (size_t)wave * T_ + tbase + r + 8 * h;
#pragma unroll
    for (int nt = 0; nt < 4; ++nt)
      pacc[row * HS_ + nt * 16 + ln] = acc[nt][r];
    if (ln == 0) { pm[row] = mrow[r]; pl[row] = lrow[r]; }
  }
}

// ---------------------------------------------------------------------------
// Kernel 3: merge the SPLIT_V partial flash results.
// out = sum_s acc_s * e^(m_s - M) / sum_s l_s * e^(m_s - M),  M = max_s m_s
// ---------------------------------------------------------------------------
__global__ __launch_bounds__(256) void merge_kernel(
    const float* __restrict__ pacc, const float* __restrict__ pm,
    const float* __restrict__ pl, float* __restrict__ out) {
  const size_t idx = (size_t)blockIdx.x * 256 + threadIdx.x;  // over B*T*HS
  const size_t row = idx / HS_;                               // = b*T + t
  const size_t stride_r = (size_t)B_ * T_;

  float M = -3.0e38f;
#pragma unroll
  for (int s = 0; s < SPLIT_V; ++s) M = fmaxf(M, pm[s * stride_r + row]);
  float num = 0.0f, den = 0.0f;
#pragma unroll
  for (int s = 0; s < SPLIT_V; ++s) {
    const float w = __expf(pm[s * stride_r + row] - M);
    num += pacc[s * stride_r * HS_ + idx] * w;
    den += pl[s * stride_r + row] * w;
  }
  out[idx] = num / den;
}

// ---------------------------------------------------------------------------
extern "C" void kernel_launch(void* const* d_in, const int* in_sizes, int n_in,
                              void* d_out, int out_size, void* d_ws,
                              size_t ws_size, hipStream_t stream) {
  const float* x   = (const float*)d_in[0];
  const float* Wk  = (const float*)d_in[1];
  const float* bk  = (const float*)d_in[2];
  const float* Wq  = (const float*)d_in[3];
  const float* bq  = (const float*)d_in[4];
  const float* Wv  = (const float*)d_in[5];
  const float* bv  = (const float*)d_in[6];
  const float* rel = (const float*)d_in[7];
  float* out = (float*)d_out;

  const size_t nqkv = (size_t)B_ * T_ * HS_;   // 1M elements
  __bf16* qB  = (__bf16*)d_ws;                 // [B,T,HS]          2 MB
  __bf16* kB  = qB + nqkv;                     // [B,T,HS] (x8)     2 MB
  __bf16* vBt = kB + nqkv;                     // [B,HS,T]          2 MB
  float* pacc = (float*)(vBt + nqkv);          // [S,B,T,HS]       16 MB
  float* pm   = pacc + (size_t)SPLIT_V * nqkv; // [S,B,T]         256 KB
  float* pl   = pm + (size_t)SPLIT_V * B_ * T_;

  const dim3 g1((B_ * T_) / 16), b1(128);
  proj_kernel<false><<<g1, b1, 0, stream>>>(x, Wq, bq, qB, 1.0f);
  proj_kernel<false><<<g1, b1, 0, stream>>>(x, Wk, bk, kB, 8.0f);  // sqrt(HS)=8
  proj_kernel<true ><<<g1, b1, 0, stream>>>(x, Wv, bv, vBt, 1.0f);

  attn_kernel<<<dim3(T_ / 16, SPLIT_V), dim3(256), 0, stream>>>(
      qB, kB, vBt, rel, pacc, pm, pl);

  merge_kernel<<<dim3((B_ * T_ * HS_) / 256), dim3(256), 0, stream>>>(
      pacc, pm, pl, out);
}
